// MessageAggregator_91027536872092
// MI455X (gfx1250) — matrix operand order repaired and checked by reading, not compiled, not run on hardware
//
#include <hip/hip_runtime.h>
#include <hip/hip_bf16.h>

// ---------------------------------------------------------------------------
// Fused temporal-graph message aggregation for MI455X (gfx1250, wave32, WMMA).
// 4 targets per 256-thread block (128 rows). Each wave owns a 16-wide column
// tile with 8 row-tile accumulators; A and B fragments are software-
// prefetched so DS/VMEM latency overlaps the v_wmma_f32_16x16x32_bf16 pipe.
// V is stored transposed (VT[hd][s]) so the P@V B-fragments are two
// ds_load_b128 instead of 16 scalar loads.
// ---------------------------------------------------------------------------

typedef __attribute__((ext_vector_type(16))) __bf16          v16bf;
typedef __attribute__((ext_vector_type(8)))  float           v8f;
typedef __attribute__((ext_vector_type(16))) unsigned short  v16u;

#define GRP   32
#define NFD   128
#define EFD   64
#define DM    128
#define XK    320
#define NH    4
#define HDIM  32
#define TGB   4
#define RB    (TGB * GRP)   // 128 rows per block

// LDS strides (elements)
#define XS    328
#define HS    136
#define QS    392
#define PS    40
#define SCS   33
#define FS    132
#define VTS   40

// LDS regions (bytes); aliased by lifetime
#define OFF_X 0          // X: 128*328*2=83968 | SC: 16*32*33*4=67584 | FF: 128*132*4=67584
#define OFF_H 83968      // H: 128*136*2=34816 | O
#define OFF_M 118784     // MS: 34816          | P: 16*32*40*2=40960
#define OFF_Q 159744     // QKV: 128*392*2=100352 (only q,k sections used)
#define OFF_I 260096     // SIDS/TIDS: 2*128*4=1024
#define OFF_V 261120     // VT: 4 targets * 128 hd * 40 * 2 = 40960
#define SMEM_BYTES 302080

__device__ __forceinline__ unsigned short f2bf(float f) {
  unsigned int u = __float_as_uint(f);
  u += 0x7fffu + ((u >> 16) & 1u);        // round-to-nearest-even
  return (unsigned short)(u >> 16);
}

__device__ __forceinline__ v8f wmma_bf16(v16bf a, v16bf b, v8f c) {
  return __builtin_amdgcn_wmma_f32_16x16x32_bf16(
      false, a, false, b, (short)0, c, false, false);
}

// A fragment, 16x32 bf16 from LDS row-major. lane l: M=row0+(l&15);
// element e=g*8+i <-> K = k0 + (l>>4)*16 + g*8 + i
__device__ __forceinline__ v16bf load_a_lds(const unsigned short* base, int ld,
                                            int row0, int k0, int lane) {
  const unsigned short* p = base + (row0 + (lane & 15)) * ld + k0 + ((lane >> 4) << 4);
  union { v16u u; v16bf b; } r;
#pragma unroll
  for (int g = 0; g < 2; ++g)
#pragma unroll
    for (int i = 0; i < 8; ++i)
      r.u[g * 8 + i] = p[g * 8 + i];
  return r.b;
}

// B fragment, 32x16 bf16 with B[k][n] = W[col0+n][k]  (contiguous per lane).
__device__ __forceinline__ v16bf load_bT(const unsigned short* W, int ld,
                                         int col0, int k0, int lane) {
  const unsigned short* p = W + (col0 + (lane & 15)) * ld + k0 + ((lane >> 4) << 4);
  union { v16u u; v16bf b; } r;
#pragma unroll
  for (int i = 0; i < 16; ++i) r.u[i] = p[i];
  return r.b;
}

// Column-tile GEMM with A/B prefetch chains: wave owns cols [ct*16, ct*16+16)
// over 8 row tiles; next fragment is loaded before the current one is consumed.
__device__ __forceinline__ void gemm_col8(const unsigned short* Ab, int lda,
                                          const unsigned short* W, int ldw,
                                          int ct, int K, int lane, v8f acc[8]) {
#pragma unroll
  for (int i = 0; i < 8; ++i) acc[i] = v8f{};
  v16bf bcur = load_bT(W, ldw, ct * 16, 0, lane);
  v16bf acur = load_a_lds(Ab, lda, 0, 0, lane);
#pragma unroll 1
  for (int k0 = 0; k0 < K; k0 += 32) {
    const bool more = (k0 + 32 < K);
    v16bf bnxt = bcur;
    if (more) bnxt = load_bT(W, ldw, ct * 16, k0 + 32, lane);
#pragma unroll
    for (int rt = 0; rt < 8; ++rt) {
      v16bf anxt = acur;
      if (rt < 7)       anxt = load_a_lds(Ab, lda, (rt + 1) * 16, k0, lane);
      else if (more)    anxt = load_a_lds(Ab, lda, 0, k0 + 32, lane);
      acc[rt] = wmma_bf16(acur, bcur, acc[rt]);
      acur = anxt;
    }
    bcur = bnxt;
  }
}

__global__ void convert_f32_to_bf16(const float* __restrict__ s,
                                    unsigned short* __restrict__ d, int n) {
  int i = blockIdx.x * blockDim.x + threadIdx.x;
  if (i < n) d[i] = f2bf(s[i]);
}

__global__ __launch_bounds__(256)
void fused_msg_attn(const int* __restrict__ src, const int* __restrict__ tgt,
                    const float* __restrict__ edge_f, const float* __restrict__ node_f,
                    const unsigned short* __restrict__ W1b, const float* __restrict__ b1,
                    const unsigned short* __restrict__ W2b, const float* __restrict__ b2,
                    const unsigned short* __restrict__ IPb, const float* __restrict__ ipb,
                    const unsigned short* __restrict__ OPb, const float* __restrict__ opb,
                    float* __restrict__ out) {
  __shared__ alignas(16) unsigned char smem[SMEM_BYTES];
  unsigned short* X   = (unsigned short*)(smem + OFF_X);
  float*          SC  = (float*)         (smem + OFF_X);   // aliases X  (post stage 2)
  float*          FF  = (float*)         (smem + OFF_X);   // aliases SC (post softmax)
  unsigned short* Hh  = (unsigned short*)(smem + OFF_H);
  unsigned short* Oo  = (unsigned short*)(smem + OFF_H);   // aliases Hh (post stage 3)
  unsigned short* MS  = (unsigned short*)(smem + OFF_M);
  unsigned short* PSM = (unsigned short*)(smem + OFF_M);   // aliases MS (post stage 4)
  unsigned short* QKV = (unsigned short*)(smem + OFF_Q);
  int*            IDS = (int*)           (smem + OFF_I);
  unsigned short* VT  = (unsigned short*)(smem + OFF_V);

  const int tid  = threadIdx.x;
  const int lane = tid & 31;
  const int wave = tid >> 5;
  const int e0   = blockIdx.x * RB;

  __builtin_prefetch(W1b, 0, 1);
  __builtin_prefetch(IPb, 0, 1);

  // ---- stage 0: cache per-row gather bases in LDS ----
  if (tid < RB)            IDS[tid] = src[e0 + tid] * NFD;
  else if (tid < 2 * RB)   IDS[tid] = tgt[e0 + (tid - RB)] * NFD;
  __syncthreads();

  // ---- stage 1: gather x = [src_feat | tgt_feat | edge_feat] -> LDS bf16 ----
#pragma unroll 1
  for (int idx = tid; idx < RB * XK; idx += 256) {
    const int r = idx / XK;
    const int c = idx - r * XK;
    float v;
    if (c < NFD)            v = node_f[IDS[r] + c];
    else if (c < 2 * NFD)   v = node_f[IDS[RB + r] + (c - NFD)];
    else                    v = edge_f[(e0 + r) * EFD + (c - 2 * NFD)];
    X[r * XS + c] = f2bf(v);
  }
  __syncthreads();

  const int nl = lane & 15;
  const int m8 = (lane >> 4) << 3;

  // ---- stage 2: h = relu(x @ W1^T + b1) ----
  {
    v8f acc[8];
    gemm_col8(X, XS, W1b, XK, wave, XK, lane, acc);
    const int n  = wave * 16 + nl;
    const float bv = b1[n];
#pragma unroll
    for (int rt = 0; rt < 8; ++rt)
#pragma unroll
      for (int r = 0; r < 8; ++r) {
        float v = acc[rt][r] + bv;
        Hh[(rt * 16 + m8 + r) * HS + n] = f2bf(v > 0.f ? v : 0.f);
      }
  }
  __syncthreads();

  // ---- stage 3: msgs = h @ W2^T + b2 ----
  {
    v8f acc[8];
    gemm_col8(Hh, HS, W2b, DM, wave, DM, lane, acc);
    const int n  = wave * 16 + nl;
    const float bv = b2[n];
#pragma unroll
    for (int rt = 0; rt < 8; ++rt)
#pragma unroll
      for (int r = 0; r < 8; ++r)
        MS[(rt * 16 + m8 + r) * HS + n] = f2bf(acc[rt][r] + bv);
  }
  __syncthreads();

  // ---- stage 4: qkv = msgs @ in_proj^T + in_proj_b  (24 col tiles, 3/wave)
  //      q,k -> QKV row-major; v -> VT transposed (per target: VT[hd][s]) ----
#pragma unroll 1
  for (int j = 0; j < 3; ++j) {
    const int ct = wave + j * 8;
    v8f acc[8];
    gemm_col8(MS, HS, IPb, DM, ct, DM, lane, acc);
    const int n  = ct * 16 + nl;
    const float bv = ipb[n];
    if (j < 2) {
#pragma unroll
      for (int rt = 0; rt < 8; ++rt)
#pragma unroll
        for (int r = 0; r < 8; ++r)
          QKV[(rt * 16 + m8 + r) * QS + n] = f2bf(acc[rt][r] + bv);
    } else {
      const int hdc = (ct - 16) * 16 + nl;        // 0..127
#pragma unroll
      for (int rt = 0; rt < 8; ++rt) {
        const int row = rt * 16 + m8;             // 8-aligned: one target block
        const int tg  = row >> 5, s = row & 31;
        union { unsigned short h[8]; uint4 q; } pk;
#pragma unroll
        for (int r = 0; r < 8; ++r) pk.h[r] = f2bf(acc[rt][r] + bv);
        *(uint4*)(VT + (tg * DM + hdc) * VTS + s) = pk.q;
      }
    }
  }
  __syncthreads();

  // ---- stage 5a: S = q k^T per (target, head); 64 tiles, 8 per wave ----
#pragma unroll 1
  for (int i = 0; i < 8; ++i) {
    const int tt = wave + i * 8;          // 0..63
    const int tg = tt >> 4, hd = (tt >> 2) & 3, rt = tt & 1, ct = (tt >> 1) & 1;
    v8f c = wmma_bf16(
        load_a_lds(QKV, QS, tg * GRP + rt * 16, hd * HDIM, lane),
        load_bT(QKV + DM, QS, tg * GRP + ct * 16, hd * HDIM, lane), v8f{});
    float* sc = SC + (tg * NH + hd) * (GRP * SCS);
    const int n = ct * 16 + nl;
#pragma unroll
    for (int r = 0; r < 8; ++r)
      sc[(rt * 16 + m8 + r) * SCS + n] = c[r];
  }
  __syncthreads();

  // ---- stage 5b: softmax rows -> probs bf16 (packed stores); 512 rows ----
#pragma unroll 1
  for (int i = 0; i < 2; ++i) {
    const int idx = tid + i * 256;
    const int p   = idx >> 5, row = idx & 31;
    const float scale = 0.17677669529663687f;     // 1/sqrt(32)
    const float* srow = SC + p * (GRP * SCS) + row * SCS;
    float mx = -3.0e38f;
#pragma unroll
    for (int j = 0; j < GRP; ++j) { float v = srow[j] * scale; mx = v > mx ? v : mx; }
    float sum = 0.f;
#pragma unroll
    for (int j = 0; j < GRP; ++j) sum += __expf(srow[j] * scale - mx);
    const float inv = 1.f / sum;
    unsigned short* prow = PSM + p * (GRP * PS) + row * PS;
#pragma unroll
    for (int j0 = 0; j0 < GRP; j0 += 8) {
      union { unsigned short h[8]; uint4 q; } pk;
#pragma unroll
      for (int j = 0; j < 8; ++j)
        pk.h[j] = f2bf(__expf(srow[j0 + j] * scale - mx) * inv);
      *(uint4*)(prow + j0) = pk.q;
    }
  }
  __syncthreads();

  // ---- stage 6: o = P v per (target, head); B from transposed VT ----
#pragma unroll 1
  for (int i = 0; i < 8; ++i) {
    const int tt = wave + i * 8;
    const int tg = tt >> 4, hd = (tt >> 2) & 3, rt = tt & 1, ct = (tt >> 1) & 1;
    v8f c = wmma_bf16(
        load_a_lds(PSM + (tg * NH + hd) * (GRP * PS), PS, rt * 16, 0, lane),
        load_bT(VT + tg * DM * VTS, VTS, hd * HDIM + ct * 16, 0, lane), v8f{});
    const int n = hd * HDIM + ct * 16 + nl;
#pragma unroll
    for (int r = 0; r < 8; ++r)
      Oo[(tg * GRP + rt * 16 + m8 + r) * HS + n] = f2bf(c[r]);
  }
  __syncthreads();

  // ---- stage 7: f = o @ out_proj^T + out_proj_b ----
  {
    v8f acc[8];
    gemm_col8(Oo, HS, OPb, DM, wave, DM, lane, acc);
    const int n  = wave * 16 + nl;
    const float bv = opb[n];
#pragma unroll
    for (int rt = 0; rt < 8; ++rt)
#pragma unroll
      for (int r = 0; r < 8; ++r)
        FF[(rt * 16 + m8 + r) * FS + n] = acc[rt][r] + bv;
  }
  __syncthreads();

  // ---- stage 8: per-target mean over 32 rows; 512 outputs ----
#pragma unroll 1
  for (int i = 0; i < 2; ++i) {
    const int idx = tid + i * 256;
    const int tg  = idx >> 7, col = idx & 127;
    float s = 0.f;
#pragma unroll
    for (int r = 0; r < GRP; ++r) s += FF[(tg * GRP + r) * FS + col];
    out[(blockIdx.x * TGB + tg) * DM + col] = s * (1.f / (float)GRP);
  }
}

extern "C" void kernel_launch(void* const* d_in, const int* in_sizes, int n_in,
                              void* d_out, int out_size, void* d_ws, size_t ws_size,
                              hipStream_t stream) {
  const int*   src    = (const int*)  d_in[0];
  const int*   tgt    = (const int*)  d_in[1];
  const float* edge_f = (const float*)d_in[2];
  const float* node_f = (const float*)d_in[3];
  // d_in[4] = timestamps (unused by the reference math)
  const float* W1  = (const float*)d_in[5];
  const float* b1  = (const float*)d_in[6];
  const float* W2  = (const float*)d_in[7];
  const float* b2  = (const float*)d_in[8];
  const float* IP  = (const float*)d_in[9];
  const float* ipb = (const float*)d_in[10];
  const float* OP  = (const float*)d_in[11];
  const float* opb = (const float*)d_in[12];
  float* out = (float*)d_out;

  const int E  = in_sizes[0];
  const int nT = E / GRP;
  const int nB = nT / TGB;

  const int nW1 = DM * XK;        // 40960
  const int nW2 = DM * DM;        // 16384
  const int nIP = 3 * DM * DM;    // 49152
  const int nOP = DM * DM;        // 16384
  unsigned short* ws   = (unsigned short*)d_ws;
  unsigned short* W1b  = ws;
  unsigned short* W2b  = ws + nW1;
  unsigned short* IPb  = ws + nW1 + nW2;
  unsigned short* OPb  = ws + nW1 + nW2 + nIP;

  convert_f32_to_bf16<<<(nW1 + 255) / 256, 256, 0, stream>>>(W1, W1b, nW1);
  convert_f32_to_bf16<<<(nW2 + 255) / 256, 256, 0, stream>>>(W2, W2b, nW2);
  convert_f32_to_bf16<<<(nIP + 255) / 256, 256, 0, stream>>>(IP, IPb, nIP);
  convert_f32_to_bf16<<<(nOP + 255) / 256, 256, 0, stream>>>(OP, OPb, nOP);

  fused_msg_attn<<<nB, 256, 0, stream>>>(src, tgt, edge_f, node_f,
                                         W1b, b1, W2b, b2, IPb, ipb, OPb, opb, out);
}
